// TFBlipAttention_72370198937788
// MI455X (gfx1250) — matrix operand order repaired and checked by reading, not compiled, not run on hardware
//
#include <hip/hip_runtime.h>

// MI455X / gfx1250 BLIP attention block:
//   QKV GEMM (f16 WMMA) -> flash attention (f16 WMMA + f32 online softmax) -> proj GEMM (f16 WMMA)
// Scratch (d_ws): q[B,H,T,D] f16 | k[B,H,T,D] f16 | vT[B,H,D,T] f16 | ctx[B,T,C] f16
// CDNA5 paths: v_wmma_f32_16x16x32_f16, global_load_async_to_lds_b128 (+ s_wait_asynccnt),
//              ds_load_b128 fragment feeds, s_wait_dscnt-synced per-wave P re-layout.

typedef __attribute__((ext_vector_type(16))) _Float16 v16h;
typedef __attribute__((ext_vector_type(8)))  _Float16 v8h;
typedef __attribute__((ext_vector_type(8)))  float    v8f;

#define WMMA_F16(a, b, c) \
  __builtin_amdgcn_wmma_f32_16x16x32_f16(false, (a), false, (b), (short)0, (c), false, false)

static constexpr int Bc = 8, Tc = 1024, Cc = 768, Hc = 12, Dc = 64;
static constexpr int MROWS = Bc * Tc;   // 8192
static constexpr int NQKV  = 3 * Cc;    // 2304
static constexpr float ATT_SCALE = 0.125f;  // 64^-0.5

// ---- CDNA5 async global->LDS copy (ASYNCcnt-tracked, VGPR-bypassing) --------
__device__ __forceinline__ void async_g2l_b128(unsigned lds_byte_addr,
                                               unsigned long long gaddr) {
  asm volatile("global_load_async_to_lds_b128 %0, %1, off"
               :: "v"(lds_byte_addr), "v"(gaddr) : "memory");
}
__device__ __forceinline__ void wait_async0() {
  asm volatile("s_wait_asynccnt 0" ::: "memory");
}
__device__ __forceinline__ unsigned lds_addr32(const void* p) {
  // generic LDS address: addr[31:0] == LDS byte offset (CDNA5 aperture rule)
  return (unsigned)(uintptr_t)p;
}

// ---- WMMA fragment loaders (layouts per CDNA5 ISA 7.12.2) --------------------
// A 16x32 f16: lane row m = lane&15, group g = lane>>4; element e -> K = 16*(e/8) + 8g + (e&7)
__device__ __forceinline__ v16h frag_a_from(const _Float16* p_row, int g) {
  const v8h lo = *(const v8h*)(p_row + 8 * g);
  const v8h hi = *(const v8h*)(p_row + 16 + 8 * g);
  v16h r;
#pragma unroll
  for (int i = 0; i < 8; ++i) { r[i] = lo[i]; r[i + 8] = hi[i]; }
  return r;
}
// B 32x16 f16: lane col n = lane&15, group g = lane>>4; element e -> K = 16*(e/8) + 8g + (e&7)
__device__ __forceinline__ v16h frag_b_from(const _Float16* p_col, int g) {
  const v8h lo = *(const v8h*)(p_col + 8 * g);
  const v8h hi = *(const v8h*)(p_col + 16 + 8 * g);
  v16h r;
#pragma unroll
  for (int i = 0; i < 8; ++i) { r[i] = lo[i]; r[i + 8] = hi[i]; }
  return r;
}

// ---- Kernel 1: QKV GEMM: X[8192,768]f32 x W[768,2304]f32 + b -> q,k f16 [B,H,T,D]; vT f16 [B,H,D,T]
__global__ __launch_bounds__(256) void qkv_gemm_kernel(
    const float* __restrict__ X, const float* __restrict__ W, const float* __restrict__ bias,
    _Float16* __restrict__ qb, _Float16* __restrict__ kb, _Float16* __restrict__ vtb)
{
  __shared__ __align__(16) _Float16 As[128 * 32];   // [m][k]
  __shared__ __align__(16) _Float16 Bs[128 * 32];   // [n][k] (transposed)
  const int tid  = threadIdx.x;
  const int lane = tid & 31, wid = tid >> 5;
  const int g = lane >> 4, ln = lane & 15;
  const int wm = (wid >> 2) * 64;        // 0 or 64
  const int wn = (wid & 3) * 32;         // 0,32,64,96
  const int m0 = blockIdx.y * 128;
  const int n0 = blockIdx.x * 128;

  const v8f vzero = {0.f, 0.f, 0.f, 0.f, 0.f, 0.f, 0.f, 0.f};
  v8f acc[4][2];
#pragma unroll
  for (int i = 0; i < 4; ++i)
#pragma unroll
    for (int j = 0; j < 2; ++j) acc[i][j] = vzero;

  for (int k0 = 0; k0 < Cc; k0 += 32) {
    // stage A: 128x32, f32 -> f16, packed to two ds_store_b128 per thread
    {
      const int idx = tid * 16;
      const int r = idx >> 5, kk = idx & 31;   // kk in {0,16}
      const float* src = X + (size_t)(m0 + r) * Cc + k0 + kk;
      v8h h0, h1;
#pragma unroll
      for (int u = 0; u < 8; ++u) { h0[u] = (_Float16)src[u]; h1[u] = (_Float16)src[u + 8]; }
      *(v8h*)(As + r * 32 + kk)     = h0;
      *(v8h*)(As + r * 32 + kk + 8) = h1;
    }
    // stage B transposed: Bs[n][kk] = W[(k0+kk)*2304 + n0+n]  (coalesced global reads)
#pragma unroll
    for (int u = 0; u < 16; ++u) {
      const int idx = tid + u * 256;
      const int kk = idx >> 7, n = idx & 127;
      Bs[n * 32 + kk] = (_Float16)W[(size_t)(k0 + kk) * NQKV + n0 + n];
    }
    __syncthreads();

    const v16h bf0 = frag_b_from(Bs + (wn + ln) * 32, g);
    const v16h bf1 = frag_b_from(Bs + (wn + 16 + ln) * 32, g);
#pragma unroll
    for (int i = 0; i < 4; ++i) {
      const v16h af = frag_a_from(As + (wm + i * 16 + ln) * 32, g);
      acc[i][0] = WMMA_F16(af, bf0, acc[i][0]);
      acc[i][1] = WMMA_F16(af, bf1, acc[i][1]);
    }
    __syncthreads();
  }

  // epilogue: bias + scatter to q/k/vT (f16)
#pragma unroll
  for (int i = 0; i < 4; ++i)
#pragma unroll
    for (int j = 0; j < 2; ++j) {
      const int c = n0 + wn + j * 16 + ln;
      const float bv = bias[c];
      const int s  = c / Cc;          // 0=q 1=k 2=v (uniform within 16-col frag)
      const int cc = c - s * Cc;
      const int h = cc >> 6, d = cc & 63;
#pragma unroll
      for (int r = 0; r < 8; ++r) {
        const int m = m0 + wm + i * 16 + r + 8 * g;
        const int b = m >> 10, t = m & 1023;
        const int bh = b * Hc + h;
        const _Float16 hv = (_Float16)(acc[i][j][r] + bv);
        if (s == 0)      qb[((size_t)bh * Tc + t) * Dc + d] = hv;
        else if (s == 1) kb[((size_t)bh * Tc + t) * Dc + d] = hv;
        else             vtb[((size_t)bh * Dc + d) * Tc + t] = hv;
      }
    }
}

// ---- Kernel 2: flash attention. 4 waves/block, 16 queries per wave, K/V tiles
//      async-staged to LDS once per block (shared by all 4 waves).
__global__ __launch_bounds__(128) void attn_kernel(
    const _Float16* __restrict__ q, const _Float16* __restrict__ k,
    const _Float16* __restrict__ vt, _Float16* __restrict__ ctx)
{
  __shared__ __align__(16) _Float16 Ks[32 * 64];         // [key][d]
  __shared__ __align__(16) _Float16 Vs[64 * 32];         // [d][key]
  __shared__ __align__(16) _Float16 Plds[4 * 16 * 32];   // per-wave P staging
  const int tid  = threadIdx.x;
  const int lane = tid & 31, wid = tid >> 5;
  const int g = lane >> 4, ln = lane & 15;
  const int bh = blockIdx.y;
  const int b = bh / Hc, h = bh - b * Hc;
  const int qbase = blockIdx.x * 64 + wid * 16;

  const _Float16* qp = q  + (size_t)bh * Tc * Dc;
  const _Float16* kp = k  + (size_t)bh * Tc * Dc;
  const _Float16* vp = vt + (size_t)bh * Dc * Tc;
  _Float16* pl = Plds + wid * 512;

  // Q fragments: 16 queries x 64 d, split K=32 twice (direct global b128 loads)
  const v16h a0 = frag_a_from(qp + (size_t)(qbase + ln) * Dc + 0,  g);
  const v16h a1 = frag_a_from(qp + (size_t)(qbase + ln) * Dc + 32, g);

  const v8f vzero = {0.f, 0.f, 0.f, 0.f, 0.f, 0.f, 0.f, 0.f};
  v8f ctxacc[4];
#pragma unroll
  for (int nd = 0; nd < 4; ++nd) ctxacc[nd] = vzero;
  float run_max[8], run_sum[8];
#pragma unroll
  for (int r = 0; r < 8; ++r) { run_max[r] = -3.0e38f; run_sum[r] = 0.f; }

  const unsigned ks_base = lds_addr32(Ks);
  const unsigned vs_base = lds_addr32(Vs);

  for (int j = 0; j < Tc; j += 32) {
    // async stage: K tile = contiguous 4KB; V^T tile = 64 rows x 64B (stride 2KB)
    {
      const unsigned long long kg = (unsigned long long)(uintptr_t)(kp + (size_t)j * Dc);
      async_g2l_b128(ks_base + (unsigned)(tid * 16),         kg + (unsigned)(tid * 16));
      async_g2l_b128(ks_base + (unsigned)((tid + 128) * 16), kg + (unsigned)((tid + 128) * 16));
      const unsigned long long vg = (unsigned long long)(uintptr_t)(vp + j);
      const int c0 = tid, c1 = tid + 128;              // 256 x 16B chunks
      async_g2l_b128(vs_base + (unsigned)(c0 * 16),
                     vg + (unsigned)(((c0 >> 2) * Tc + (c0 & 3) * 8) * 2));
      async_g2l_b128(vs_base + (unsigned)(c1 * 16),
                     vg + (unsigned)(((c1 >> 2) * Tc + (c1 & 3) * 8) * 2));
    }
    wait_async0();
    __syncthreads();

    // scores S = Q x K^T for 16 queries x 32 keys (2 groups of 16 keys)
    const v16h kb00 = frag_b_from(Ks + (ln) * 64 + 0,       g);
    const v16h kb01 = frag_b_from(Ks + (ln) * 64 + 32,      g);
    const v16h kb10 = frag_b_from(Ks + (16 + ln) * 64 + 0,  g);
    const v16h kb11 = frag_b_from(Ks + (16 + ln) * 64 + 32, g);
    v8f s0 = vzero, s1 = vzero;
    s0 = WMMA_F16(a0, kb00, s0);
    s0 = WMMA_F16(a1, kb01, s0);
    s1 = WMMA_F16(a0, kb10, s1);
    s1 = WMMA_F16(a1, kb11, s1);

    // online softmax over this 32-key tile (rows r+8g; cols across 16-lane group)
    float corr[8];
#pragma unroll
    for (int r = 0; r < 8; ++r) {
      const float x0 = s0[r] * ATT_SCALE;
      const float x1 = s1[r] * ATT_SCALE;
      float mt = fmaxf(x0, x1);
      mt = fmaxf(mt, __shfl_xor(mt, 1, 32));
      mt = fmaxf(mt, __shfl_xor(mt, 2, 32));
      mt = fmaxf(mt, __shfl_xor(mt, 4, 32));
      mt = fmaxf(mt, __shfl_xor(mt, 8, 32));
      const float nm = fmaxf(run_max[r], mt);
      const float cf = __expf(run_max[r] - nm);
      run_max[r] = nm;
      corr[r] = cf;
      const float p0 = __expf(x0 - nm);
      const float p1 = __expf(x1 - nm);
      s0[r] = p0; s1[r] = p1;
      float ps = p0 + p1;
      ps += __shfl_xor(ps, 1, 32);
      ps += __shfl_xor(ps, 2, 32);
      ps += __shfl_xor(ps, 4, 32);
      ps += __shfl_xor(ps, 8, 32);
      run_sum[r] = run_sum[r] * cf + ps;
    }
#pragma unroll
    for (int nd = 0; nd < 4; ++nd)
#pragma unroll
      for (int r = 0; r < 8; ++r) ctxacc[nd][r] *= corr[r];

    // re-layout P: D-frag (row r+8g, col ln) -> A-frag via per-wave LDS
#pragma unroll
    for (int r = 0; r < 8; ++r) {
      pl[(r + 8 * g) * 32 + ln]      = (_Float16)s0[r];
      pl[(r + 8 * g) * 32 + 16 + ln] = (_Float16)s1[r];
    }
    asm volatile("s_wait_dscnt 0" ::: "memory");
    const v16h pf = frag_a_from(pl + ln * 32, g);

    // ctx += P x V  (V^T LDS tile: rows d, contiguous over keys = contraction dim)
#pragma unroll
    for (int nd = 0; nd < 4; ++nd) {
      const v16h vf = frag_b_from(Vs + (nd * 16 + ln) * 32, g);
      ctxacc[nd] = WMMA_F16(pf, vf, ctxacc[nd]);
    }
    __syncthreads();   // protect K/V tiles before next iteration's async writes
  }

  // finalize: divide by row sum, write ctx f16 in [B,T,C] layout
  float inv[8];
#pragma unroll
  for (int r = 0; r < 8; ++r) inv[r] = 1.0f / run_sum[r];
#pragma unroll
  for (int nd = 0; nd < 4; ++nd)
#pragma unroll
    for (int r = 0; r < 8; ++r) {
      const int t = qbase + r + 8 * g;
      const int c = h * Dc + nd * 16 + ln;
      ctx[((size_t)(b * Tc + t)) * Cc + c] = (_Float16)(ctxacc[nd][r] * inv[r]);
    }
}

// ---- Kernel 3: projection GEMM: ctx[8192,768]f16 x W[768,768]f32 + b -> out f32
__global__ __launch_bounds__(256) void proj_gemm_kernel(
    const _Float16* __restrict__ Xh, const float* __restrict__ W, const float* __restrict__ bias,
    float* __restrict__ out)
{
  __shared__ __align__(16) _Float16 As[128 * 32];
  __shared__ __align__(16) _Float16 Bs[128 * 32];   // [n][k]
  const int tid  = threadIdx.x;
  const int lane = tid & 31, wid = tid >> 5;
  const int g = lane >> 4, ln = lane & 15;
  const int wm = (wid >> 2) * 64;
  const int wn = (wid & 3) * 32;
  const int m0 = blockIdx.y * 128;
  const int n0 = blockIdx.x * 128;

  const v8f vzero = {0.f, 0.f, 0.f, 0.f, 0.f, 0.f, 0.f, 0.f};
  v8f acc[4][2];
#pragma unroll
  for (int i = 0; i < 4; ++i)
#pragma unroll
    for (int j = 0; j < 2; ++j) acc[i][j] = vzero;

  for (int k0 = 0; k0 < Cc; k0 += 32) {
    // stage A (already f16): async global->LDS b128 copies, no VGPR bounce
    {
      const int idx = tid * 16;
      const int r = idx >> 5, kk = idx & 31;   // kk in {0,16}
      const unsigned long long src =
          (unsigned long long)(uintptr_t)(Xh + (size_t)(m0 + r) * Cc + k0 + kk);
      const unsigned dst = lds_addr32(As + r * 32 + kk);
      async_g2l_b128(dst,      src);
      async_g2l_b128(dst + 16, src + 16);
    }
    // stage B transposed with f32->f16 convert (coalesced global reads)
#pragma unroll
    for (int u = 0; u < 16; ++u) {
      const int idx = tid + u * 256;
      const int kk = idx >> 7, n = idx & 127;
      Bs[n * 32 + kk] = (_Float16)W[(size_t)(k0 + kk) * Cc + n0 + n];
    }
    wait_async0();
    __syncthreads();

    const v16h bf0 = frag_b_from(Bs + (wn + ln) * 32, g);
    const v16h bf1 = frag_b_from(Bs + (wn + 16 + ln) * 32, g);
#pragma unroll
    for (int i = 0; i < 4; ++i) {
      const v16h af = frag_a_from(As + (wm + i * 16 + ln) * 32, g);
      acc[i][0] = WMMA_F16(af, bf0, acc[i][0]);
      acc[i][1] = WMMA_F16(af, bf1, acc[i][1]);
    }
    __syncthreads();
  }

#pragma unroll
  for (int i = 0; i < 4; ++i)
#pragma unroll
    for (int j = 0; j < 2; ++j) {
      const int c = n0 + wn + j * 16 + ln;
      const float bv = bias[c];
#pragma unroll
      for (int r = 0; r < 8; ++r) {
        const int m = m0 + wm + i * 16 + r + 8 * g;
        out[(size_t)m * Cc + c] = acc[i][j][r] + bv;
      }
    }
}

extern "C" void kernel_launch(void* const* d_in, const int* in_sizes, int n_in,
                              void* d_out, int out_size, void* d_ws, size_t ws_size,
                              hipStream_t stream) {
  const float* X      = (const float*)d_in[0];
  const float* qkv_w  = (const float*)d_in[1];
  const float* qkv_b  = (const float*)d_in[2];
  const float* proj_w = (const float*)d_in[3];
  const float* proj_b = (const float*)d_in[4];
  float* out = (float*)d_out;

  const size_t QS = (size_t)Bc * Hc * Tc * Dc;   // 6,291,456 halves each
  _Float16* qb  = (_Float16*)d_ws;
  _Float16* kb  = qb + QS;
  _Float16* vtb = kb + QS;
  _Float16* ctx = vtb + QS;                      // [B,T,C] f16

  dim3 blk(256);
  dim3 grd1(NQKV / 128, MROWS / 128);            // 18 x 64
  qkv_gemm_kernel<<<grd1, blk, 0, stream>>>(X, qkv_w, qkv_b, qb, kb, vtb);

  dim3 blk2(128);
  dim3 grd2(Tc / 64, Bc * Hc);                   // 16 x 96
  attn_kernel<<<grd2, blk2, 0, stream>>>(qb, kb, vtb, ctx);

  dim3 grd3(Cc / 128, MROWS / 128);              // 6 x 64
  proj_gemm_kernel<<<grd3, blk, 0, stream>>>(ctx, proj_w, proj_b, out);
}
